// HybridGCNGATModel_20366734917867
// MI455X (gfx1250) — matrix operand order repaired and checked by reading, not compiled, not run on hardware
//
#include <hip/hip_runtime.h>

#define NN   20000
#define EE   320000
#define ELL  (EE + NN)
#define FIN  128
#define HID  512
#define BB   64
#define NCLS 38
#define LDT  40   // LDS row stride (ushorts) for 32-wide k tiles, padded (80B, 16B-aligned)

typedef __attribute__((ext_vector_type(16))) __bf16 v16bf;
typedef __attribute__((ext_vector_type(8)))  float  v8f;

// ---------------- helpers ----------------
__device__ inline unsigned short f2bf(float f) {
  unsigned u = __float_as_uint(f);
  u += 0x7FFFu + ((u >> 16) & 1u);          // round-to-nearest-even
  return (unsigned short)(u >> 16);
}
__device__ inline float bf2f(unsigned short h) {
  return __uint_as_float(((unsigned)h) << 16);
}
__device__ inline unsigned pack2(unsigned short a, unsigned short b) {
  return (unsigned)a | ((unsigned)b << 16);
}
__device__ inline void atomicMaxF(float* a, float v) {
  unsigned* p = (unsigned*)a;
  unsigned cur = *p;
  while (__uint_as_float(cur) < v) {
    unsigned old = atomicCAS(p, cur, __float_as_uint(v));
    if (old == cur) break;
    cur = old;
  }
}
__device__ inline float lrelu2(float x) { return x > 0.f ? x : 0.2f * x; }

// ---------------- generic fill ----------------
__global__ void fill_kernel(float* p, float v, long n) {
  long i = blockIdx.x * (long)blockDim.x + threadIdx.x;
  if (i < n) p[i] = v;
}

// ---------------- weight convert: W[K,Nc] f32 -> Wt_hi/lo[Nc,K] bf16 ----------------
__global__ void convert_w_kernel(const float* __restrict__ W,
                                 unsigned short* __restrict__ wth,
                                 unsigned short* __restrict__ wtl,
                                 int K, int Nc) {
  long i = blockIdx.x * (long)blockDim.x + threadIdx.x;
  if (i >= (long)K * Nc) return;
  int n = (int)(i / K), k = (int)(i % K);
  float v = W[(size_t)k * Nc + n];
  unsigned short h = f2bf(v);
  wth[i] = h;
  wtl[i] = f2bf(v - bf2f(h));
}

// ---------------- WMMA fragment loads (CDNA5 16-bit layouts, ISA 7.12.2) ----------------
__device__ inline v16bf frag_a(const unsigned short* row, int h) {
  // A 16x32: lane m=lane&15, h=lane>>4; elem e<8 -> K=h*8+e; e>=8 -> K=h*8+e+8
  union { v16bf v; unsigned short u[16]; } r;
  const unsigned short* p = row + h * 8;
#pragma unroll
  for (int e = 0; e < 8; e++) { r.u[e] = p[e]; r.u[e + 8] = p[e + 16]; }
  return r.v;
}
__device__ inline v16bf frag_b(const unsigned short* row, int h) {
  // B 32x16 (stored N-major in LDS): elem e<8 -> K=h*8+e; e>=8 -> K=h*8+e+8, fixed n
  union { v16bf v; unsigned short u[16]; } r;
  const unsigned short* p = row + h * 8;
#pragma unroll
  for (int e = 0; e < 8; e++) { r.u[e] = p[e]; r.u[e + 8] = p[e + 16]; }
  return r.v;
}

// ---------------- split-precision bf16 WMMA GEMM: C = A(f32) @ B(f32 via hi/lo bf16) --------
// grid: (Nc/64, ceil(M/64)); block: 256 (8 waves); each wave: 16 rows x 32 cols.
// Double-buffered LDS; B tiles DMA'd global->LDS with GLOBAL_LOAD_ASYNC_TO_LDS_B128.
__global__ void __launch_bounds__(256) gemm_bf16x3_wmma_kernel(
    const float* __restrict__ A, const unsigned short* __restrict__ Bth,
    const unsigned short* __restrict__ Btl, float* __restrict__ C,
    int M, int K, int Nc) {
  __shared__ __align__(16) unsigned short sah[2][64 * LDT];
  __shared__ __align__(16) unsigned short sal[2][64 * LDT];
  __shared__ __align__(16) unsigned short sbh[2][64 * LDT];
  __shared__ __align__(16) unsigned short sbl[2][64 * LDT];

  const int tid = threadIdx.x;
  const int m0 = blockIdx.y * 64;
  const int n0 = blockIdx.x * 64;
  const int wave = tid >> 5, lane = tid & 31;
  const int wr = wave >> 1, wc = wave & 1;
  const int lm = lane & 15, lh = lane >> 4;

  v8f acc0 = {};
  v8f acc1 = {};

  const int r  = tid >> 2;        // 0..63 : row of A tile / col of Bt tile
  const int kk = (tid & 3) * 8;   // 0,8,16,24
  const int gm = m0 + r;
  const float*          ap  = A   + (size_t)gm * K + kk;
  const unsigned short* bhp = Bth + (size_t)(n0 + r) * K + kk;
  const unsigned short* blp = Btl + (size_t)(n0 + r) * K + kk;

  // LDS byte offsets (low 32 bits of the LDS-aperture address) for async DMA dest
  const unsigned bh_lds[2] = { (unsigned)(size_t)&sbh[0][r * LDT + kk],
                               (unsigned)(size_t)&sbh[1][r * LDT + kk] };
  const unsigned bl_lds[2] = { (unsigned)(size_t)&sbl[0][r * LDT + kk],
                               (unsigned)(size_t)&sbl[1][r * LDT + kk] };

  auto load_tile = [&](int k0, int buf) {
    // B hi/lo: pure global->LDS memcpy, use the CDNA5 async-to-LDS engine (ASYNCcnt)
    asm volatile("global_load_async_to_lds_b128 %0, %1, off"
                 :: "v"(bh_lds[buf]), "v"((unsigned long long)(bhp + k0)) : "memory");
    asm volatile("global_load_async_to_lds_b128 %0, %1, off"
                 :: "v"(bl_lds[buf]), "v"((unsigned long long)(blp + k0)) : "memory");
    // A: two 16B vector loads, split fp32 -> bf16 hi/lo, two 16B LDS stores
    float4 v0 = {0.f, 0.f, 0.f, 0.f}, v1 = {0.f, 0.f, 0.f, 0.f};
    if (gm < M) {
      const float4* p = (const float4*)(ap + k0);
      v0 = p[0];
      v1 = p[1];
    }
    float fv[8] = {v0.x, v0.y, v0.z, v0.w, v1.x, v1.y, v1.z, v1.w};
    unsigned short hi[8], lo[8];
#pragma unroll
    for (int e = 0; e < 8; e++) {
      hi[e] = f2bf(fv[e]);
      lo[e] = f2bf(fv[e] - bf2f(hi[e]));
    }
    uint4 ph, pl;
    ph.x = pack2(hi[0], hi[1]); ph.y = pack2(hi[2], hi[3]);
    ph.z = pack2(hi[4], hi[5]); ph.w = pack2(hi[6], hi[7]);
    pl.x = pack2(lo[0], lo[1]); pl.y = pack2(lo[2], lo[3]);
    pl.z = pack2(lo[4], lo[5]); pl.w = pack2(lo[6], lo[7]);
    *(uint4*)&sah[buf][r * LDT + kk] = ph;
    *(uint4*)&sal[buf][r * LDT + kk] = pl;
  };

  const int nk = K >> 5;
  load_tile(0, 0);
  asm volatile("s_wait_asynccnt 0" ::: "memory");
  __syncthreads();

  for (int kt = 0; kt < nk; ++kt) {
    const int buf = kt & 1;
    if (kt + 1 < nk) load_tile((kt + 1) << 5, buf ^ 1);   // prefetch next tile into other buffer

    v16bf ah  = frag_a(&sah[buf][(16 * wr + lm) * LDT], lh);
    v16bf al  = frag_a(&sal[buf][(16 * wr + lm) * LDT], lh);
    v16bf b0h = frag_b(&sbh[buf][(32 * wc + lm) * LDT], lh);
    v16bf b0l = frag_b(&sbl[buf][(32 * wc + lm) * LDT], lh);
    v16bf b1h = frag_b(&sbh[buf][(32 * wc + 16 + lm) * LDT], lh);
    v16bf b1l = frag_b(&sbl[buf][(32 * wc + 16 + lm) * LDT], lh);

    acc0 = __builtin_amdgcn_wmma_f32_16x16x32_bf16(false, ah, false, b0h, (short)0, acc0, false, false);
    acc0 = __builtin_amdgcn_wmma_f32_16x16x32_bf16(false, al, false, b0h, (short)0, acc0, false, false);
    acc0 = __builtin_amdgcn_wmma_f32_16x16x32_bf16(false, ah, false, b0l, (short)0, acc0, false, false);
    acc1 = __builtin_amdgcn_wmma_f32_16x16x32_bf16(false, ah, false, b1h, (short)0, acc1, false, false);
    acc1 = __builtin_amdgcn_wmma_f32_16x16x32_bf16(false, al, false, b1h, (short)0, acc1, false, false);
    acc1 = __builtin_amdgcn_wmma_f32_16x16x32_bf16(false, ah, false, b1l, (short)0, acc1, false, false);

    asm volatile("s_wait_asynccnt 0" ::: "memory");   // next tile's DMA landed (per-wave)
    __syncthreads();                                  // ... and visible to all waves
  }

#pragma unroll
  for (int rr = 0; rr < 8; rr++) {
    int gm2 = m0 + 16 * wr + lh * 8 + rr;   // C layout: lane n=lm, vgpr rr -> M = lh*8+rr
    if (gm2 < M) {
      C[(size_t)gm2 * Nc + n0 + 32 * wc + lm]      = acc0[rr];
      C[(size_t)gm2 * Nc + n0 + 32 * wc + 16 + lm] = acc1[rr];
    }
  }
}

// ---------------- graph kernels ----------------
__global__ void deg_kernel(const int* __restrict__ ei, float* deg) {
  int e = blockIdx.x * blockDim.x + threadIdx.x;
  if (e < EE) atomicAdd(&deg[ei[EE + e]], 1.0f);
}
__global__ void norm_kernel(float* d) {
  int i = blockIdx.x * blockDim.x + threadIdx.x;
  if (i < NN) d[i] = rsqrtf(fmaxf(d[i], 1.0f));
}

__global__ void gcn_scatter_kernel(const float* __restrict__ h, const float* __restrict__ nrm,
                                   const int* __restrict__ ei, float* __restrict__ out) {
  int e = blockIdx.x;
  int s, d;
  if (e < EE) { s = ei[e]; d = ei[EE + e]; } else { s = d = e - EE; }
  float coef = nrm[s] * nrm[d];
  const float* hr = h + (size_t)s * HID;
  float* orow = out + (size_t)d * HID;
  for (int c = threadIdx.x; c < HID; c += blockDim.x)
    atomicAdd(&orow[c], hr[c] * coef);
}

// per-node attention dots: one wave per node
__global__ void attn_dot_kernel(const float* __restrict__ h, const float* __restrict__ a_s,
                                const float* __restrict__ a_d, float* asn, float* adn) {
  int node = blockIdx.x * 8 + (threadIdx.x >> 5);
  int lane = threadIdx.x & 31;
  if (node >= NN) return;
  const float* row = h + (size_t)node * HID;
  float s0 = 0, s1 = 0, d0 = 0, d1 = 0;
  for (int c = lane; c < 256; c += 32) {
    float h0 = row[c], h1 = row[256 + c];
    s0 += h0 * a_s[c]; s1 += h1 * a_s[256 + c];
    d0 += h0 * a_d[c]; d1 += h1 * a_d[256 + c];
  }
  for (int off = 16; off; off >>= 1) {
    s0 += __shfl_xor(s0, off); s1 += __shfl_xor(s1, off);
    d0 += __shfl_xor(d0, off); d1 += __shfl_xor(d1, off);
  }
  if (lane == 0) {
    asn[node * 2] = s0; asn[node * 2 + 1] = s1;
    adn[node * 2] = d0; adn[node * 2 + 1] = d1;
  }
}

__global__ void gat_edge_max_kernel(const float* asn, const float* adn,
                                    const int* __restrict__ ei, float* mm) {
  int e = blockIdx.x * blockDim.x + threadIdx.x;
  if (e >= ELL) return;
  int s, d;
  if (e < EE) { s = ei[e]; d = ei[EE + e]; } else { s = d = e - EE; }
#pragma unroll
  for (int hh = 0; hh < 2; hh++)
    atomicMaxF(&mm[d * 2 + hh], lrelu2(asn[s * 2 + hh] + adn[d * 2 + hh]));
}
__global__ void gat_edge_sum_kernel(const float* asn, const float* adn,
                                    const int* __restrict__ ei, const float* mm, float* ss) {
  int e = blockIdx.x * blockDim.x + threadIdx.x;
  if (e >= ELL) return;
  int s, d;
  if (e < EE) { s = ei[e]; d = ei[EE + e]; } else { s = d = e - EE; }
#pragma unroll
  for (int hh = 0; hh < 2; hh++) {
    float v = lrelu2(asn[s * 2 + hh] + adn[d * 2 + hh]);
    atomicAdd(&ss[d * 2 + hh], expf(v - mm[d * 2 + hh]));
  }
}
__global__ void gat_scatter_kernel(const float* __restrict__ h, const float* asn, const float* adn,
                                   const float* mm, const float* ss, const int* __restrict__ ei,
                                   float* __restrict__ out) {
  int e = blockIdx.x;
  int s, d;
  if (e < EE) { s = ei[e]; d = ei[EE + e]; } else { s = d = e - EE; }
  float a0 = expf(lrelu2(asn[s * 2] + adn[d * 2]) - mm[d * 2]) / ss[d * 2];
  float a1 = expf(lrelu2(asn[s * 2 + 1] + adn[d * 2 + 1]) - mm[d * 2 + 1]) / ss[d * 2 + 1];
  const float* hr = h + (size_t)s * HID;
  float* orow = out + (size_t)d * HID;
  for (int c = threadIdx.x; c < HID; c += blockDim.x)
    atomicAdd(&orow[c], hr[c] * (c < 256 ? a0 : a1));
}

// ---------------- batch-norm (bias folded in; stats via atomics) ----------------
__global__ void bn_stats_kernel(const float* __restrict__ x, const float* __restrict__ bias,
                                float* sum, float* sq, int M, int C) {
  long i = blockIdx.x * (long)blockDim.x + threadIdx.x;
  if (i >= (long)M * C) return;
  int c = (int)(i % C);
  float v = x[i] + bias[c];
  atomicAdd(&sum[c], v);
  atomicAdd(&sq[c], v * v);
}
__global__ void bn_apply_kernel(float* x, const float* __restrict__ bias,
                                const float* sum, const float* sq,
                                const float* g, const float* be, int M, int C, int act) {
  long i = blockIdx.x * (long)blockDim.x + threadIdx.x;
  if (i >= (long)M * C) return;
  int c = (int)(i % C);
  float mu = sum[c] / M;
  float var = sq[c] / M - mu * mu;
  float v = (x[i] + bias[c] - mu) * rsqrtf(fmaxf(var, 0.f) + 1e-5f) * g[c] + be[c];
  if (act == 1) v = v > 0.f ? v : 0.2f * v;            // leaky_relu(0.2)
  else if (act == 2) v = v > 0.f ? v : expf(v) - 1.f;  // elu
  x[i] = v;
}

// ---------------- pooling ----------------
__global__ void pool_acc_kernel(const float* __restrict__ h, const int* __restrict__ batch,
                                float* sums, float* maxp) {
  long i = blockIdx.x * (long)blockDim.x + threadIdx.x;
  if (i >= (long)NN * HID) return;
  int node = (int)(i >> 9), c = (int)(i & 511);
  int b = batch[node];
  float v = h[i];
  atomicAdd(&sums[b * HID + c], v);
  atomicMaxF(&maxp[b * HID + c], v);
}
__global__ void pool_cnt_kernel(const int* __restrict__ batch, float* cnt) {
  int i = blockIdx.x * blockDim.x + threadIdx.x;
  if (i < NN) atomicAdd(&cnt[batch[i]], 1.0f);
}
__global__ void pool_cat_kernel(const float* sums, const float* maxp, const float* cnt, float* p) {
  int i = blockIdx.x * blockDim.x + threadIdx.x;
  if (i >= BB * HID) return;
  int b = i >> 9, c = i & 511;
  p[b * 1024 + c]       = sums[i] / fmaxf(cnt[b], 1.0f);
  p[b * 1024 + 512 + c] = maxp[i];
}

// ---------------- final classifier layer ----------------
__global__ void out_kernel(const float* __restrict__ z, const float* __restrict__ W,
                           const float* __restrict__ b, float* __restrict__ out) {
  int i = blockIdx.x * blockDim.x + threadIdx.x;
  if (i >= BB * NCLS) return;
  int r = i / NCLS, c = i % NCLS;
  float s = b[c];
  for (int k = 0; k < 256; k++) s += z[r * 256 + k] * W[k * NCLS + c];
  out[i] = s;
}

// ---------------- host orchestration ----------------
extern "C" void kernel_launch(void* const* d_in, const int* in_sizes, int n_in,
                              void* d_out, int out_size, void* d_ws, size_t ws_size,
                              hipStream_t stream) {
  (void)in_sizes; (void)n_in; (void)out_size; (void)ws_size;
  const float* x      = (const float*)d_in[0];
  const int*   ei     = (const int*)d_in[1];
  const int*   batch  = (const int*)d_in[2];
  const float* ft_W   = (const float*)d_in[3];
  const float* ft_b   = (const float*)d_in[4];
  const float* ft_g   = (const float*)d_in[5];
  const float* ft_be  = (const float*)d_in[6];
  const float* gcn1_W = (const float*)d_in[7];
  const float* gcn1_b = (const float*)d_in[8];
  const float* gbn1_g = (const float*)d_in[9];
  const float* gbn1_b = (const float*)d_in[10];
  const float* gcn2_W = (const float*)d_in[11];
  const float* gcn2_b = (const float*)d_in[12];
  const float* gbn2_g = (const float*)d_in[13];
  const float* gbn2_b = (const float*)d_in[14];
  const float* gat1_W = (const float*)d_in[15];
  const float* gat1_as = (const float*)d_in[16];
  const float* gat1_ad = (const float*)d_in[17];
  const float* gat1_b = (const float*)d_in[18];
  const float* abn1_g = (const float*)d_in[19];
  const float* abn1_b = (const float*)d_in[20];
  const float* gat2_W = (const float*)d_in[21];
  const float* gat2_as = (const float*)d_in[22];
  const float* gat2_ad = (const float*)d_in[23];
  const float* gat2_b = (const float*)d_in[24];
  const float* abn2_g = (const float*)d_in[25];
  const float* abn2_b = (const float*)d_in[26];
  const float* fc1_W  = (const float*)d_in[27];
  const float* fc1_b  = (const float*)d_in[28];
  const float* cbn1_g = (const float*)d_in[29];
  const float* cbn1_b = (const float*)d_in[30];
  const float* fc2_W  = (const float*)d_in[31];
  const float* fc2_b  = (const float*)d_in[32];
  const float* cbn2_g = (const float*)d_in[33];
  const float* cbn2_b = (const float*)d_in[34];
  const float* out_W  = (const float*)d_in[35];
  const float* out_b  = (const float*)d_in[36];

  // workspace layout
  float* hA = (float*)d_ws;                               // [NN,HID]
  float* hB = hA + (size_t)NN * HID;                      // [NN,HID]
  unsigned short* wth = (unsigned short*)(hB + (size_t)NN * HID);  // [1024*512] bf16
  unsigned short* wtl = wth + 1024 * 512;
  float* nrm  = (float*)(wtl + 1024 * 512);               // [NN]
  float* asn  = nrm + NN;                                 // [NN,2]
  float* adn  = asn + 2 * NN;
  float* mm   = adn + 2 * NN;
  float* ss   = mm + 2 * NN;
  float* psum = ss + 2 * NN;                              // [BB,HID]
  float* pmax = psum + BB * HID;
  float* pcnt = pmax + BB * HID;                          // [BB]
  float* pcat = pcnt + BB;                                // [BB,1024]
  float* z1   = pcat + BB * 1024;                         // [BB,512]
  float* z2   = z1 + BB * HID;                            // [BB,256]
  float* bnsum = z2 + BB * 256;                           // [512]
  float* bnsq  = bnsum + HID;                             // [512]

  auto fill = [&](float* p, float v, long n) {
    fill_kernel<<<(unsigned)((n + 255) / 256), 256, 0, stream>>>(p, v, n);
  };
  auto gemm = [&](const float* A, const float* W, float* Cc, int M, int K, int Nc) {
    long kn = (long)K * Nc;
    convert_w_kernel<<<(unsigned)((kn + 255) / 256), 256, 0, stream>>>(W, wth, wtl, K, Nc);
    gemm_bf16x3_wmma_kernel<<<dim3(Nc / 64, (M + 63) / 64), 256, 0, stream>>>(A, wth, wtl, Cc, M, K, Nc);
  };
  auto bn = [&](float* data, const float* bias, const float* g, const float* be,
                int M, int C, int act) {
    fill(bnsum, 0.f, C);
    fill(bnsq, 0.f, C);
    long t = (long)M * C;
    bn_stats_kernel<<<(unsigned)((t + 255) / 256), 256, 0, stream>>>(data, bias, bnsum, bnsq, M, C);
    bn_apply_kernel<<<(unsigned)((t + 255) / 256), 256, 0, stream>>>(data, bias, bnsum, bnsq, g, be, M, C, act);
  };

  // degree normalization (shared by both GCN layers)
  fill(nrm, 1.0f, NN);                                    // self-loop
  deg_kernel<<<(EE + 255) / 256, 256, 0, stream>>>(ei, nrm);
  norm_kernel<<<(NN + 255) / 256, 256, 0, stream>>>(nrm);

  // feature transform: Linear -> BN -> LeakyReLU(0.2)
  gemm(x, ft_W, hA, NN, FIN, HID);
  bn(hA, ft_b, ft_g, ft_be, NN, HID, 1);

  // GCN block (x2): GEMM -> scatter -> BN -> ELU
  for (int l = 0; l < 2; l++) {
    const float* W  = l == 0 ? gcn1_W : gcn2_W;
    const float* b  = l == 0 ? gcn1_b : gcn2_b;
    const float* g  = l == 0 ? gbn1_g : gbn2_g;
    const float* be = l == 0 ? gbn1_b : gbn2_b;
    gemm(hA, W, hB, NN, HID, HID);
    fill(hA, 0.f, (long)NN * HID);
    gcn_scatter_kernel<<<ELL, 256, 0, stream>>>(hB, nrm, ei, hA);
    bn(hA, b, g, be, NN, HID, 2);
  }

  // GAT block (x2): GEMM -> attn dots -> segment softmax -> scatter -> BN -> ELU
  for (int l = 0; l < 2; l++) {
    const float* W  = l == 0 ? gat1_W : gat2_W;
    const float* as_ = l == 0 ? gat1_as : gat2_as;
    const float* ad_ = l == 0 ? gat1_ad : gat2_ad;
    const float* b  = l == 0 ? gat1_b : gat2_b;
    const float* g  = l == 0 ? abn1_g : abn2_g;
    const float* be = l == 0 ? abn1_b : abn2_b;
    gemm(hA, W, hB, NN, HID, HID);
    attn_dot_kernel<<<(NN + 7) / 8, 256, 0, stream>>>(hB, as_, ad_, asn, adn);
    fill(mm, -1e30f, 2 * NN);
    fill(ss, 0.f, 2 * NN);
    gat_edge_max_kernel<<<(ELL + 255) / 256, 256, 0, stream>>>(asn, adn, ei, mm);
    gat_edge_sum_kernel<<<(ELL + 255) / 256, 256, 0, stream>>>(asn, adn, ei, mm, ss);
    fill(hA, 0.f, (long)NN * HID);
    gat_scatter_kernel<<<ELL, 256, 0, stream>>>(hB, asn, adn, mm, ss, ei, hA);
    bn(hA, b, g, be, NN, HID, 2);
  }

  // global mean+max pool
  fill(psum, 0.f, BB * HID);
  fill(pmax, -1e30f, BB * HID);
  fill(pcnt, 0.f, BB);
  pool_acc_kernel<<<(unsigned)(((long)NN * HID + 255) / 256), 256, 0, stream>>>(hA, batch, psum, pmax);
  pool_cnt_kernel<<<(NN + 255) / 256, 256, 0, stream>>>(batch, pcnt);
  pool_cat_kernel<<<(BB * HID + 255) / 256, 256, 0, stream>>>(psum, pmax, pcnt, pcat);

  // classifier
  gemm(pcat, fc1_W, z1, BB, 2 * HID, HID);
  bn(z1, fc1_b, cbn1_g, cbn1_b, BB, HID, 0);
  gemm(z1, fc2_W, z2, BB, HID, 256);
  bn(z2, fc2_b, cbn2_g, cbn2_b, BB, 256, 1);
  out_kernel<<<(BB * NCLS + 255) / 256, 256, 0, stream>>>(z2, out_W, out_b, (float*)d_out);
}
